// GNNModel_49417893708345
// MI455X (gfx1250) — matrix operand order repaired and checked by reading, not compiled, not run on hardware
//
#include <hip/hip_runtime.h>
#include <math.h>

// ---------------------------------------------------------------------------
// Problem constants (from reference)
// ---------------------------------------------------------------------------
#define N_NODES_C  100000
#define N_EDGES_C  800000
#define N_GRAPHS_C 1000
#define IN_F_C     128
#define HID_C      256
#define OUT_F_C    40

// ---------------------------------------------------------------------------
// CDNA5 WMMA types
// ---------------------------------------------------------------------------
typedef __attribute__((ext_vector_type(16))) __bf16        v16bf;
typedef __attribute__((ext_vector_type(8)))  float         v8f;
typedef __attribute__((ext_vector_type(8)))  unsigned int  v8u;
typedef __attribute__((ext_vector_type(4)))  unsigned int  v4u;

__device__ __forceinline__ unsigned short f2bf(float f) {
    unsigned int u = __builtin_bit_cast(unsigned int, f);
    unsigned int r = (u + 0x7FFFu + ((u >> 16) & 1u)) >> 16;   // RNE
    return (unsigned short)r;
}

// ---------------------------------------------------------------------------
// Utility kernels
// ---------------------------------------------------------------------------
__global__ void zero_f32(float* __restrict__ p, long n) {
    long stride = (long)gridDim.x * blockDim.x;
    for (long i = (long)blockIdx.x * blockDim.x + threadIdx.x; i < n; i += stride)
        p[i] = 0.0f;
}

__global__ void deg_kernel(const int* __restrict__ dst, float* __restrict__ deg) {
    int e = blockIdx.x * blockDim.x + threadIdx.x;
    if (e < N_EDGES_C) atomicAdd(&deg[dst[e]], 1.0f);
}

__global__ void count_kernel(const int* __restrict__ bvec, int* __restrict__ cnt) {
    int i = blockIdx.x * blockDim.x + threadIdx.x;
    if (i < N_NODES_C) atomicAdd(&cnt[bvec[i]], 1);
}

__global__ void scan_bases(const int* __restrict__ cnt, int* __restrict__ base) {
    if (blockIdx.x == 0 && threadIdx.x == 0) {
        int acc = 0;
        for (int g = 0; g < N_GRAPHS_C; ++g) { base[g] = acc; acc += cnt[g]; }
    }
}

// Convert [Wl | Wr] (each HID x F, f32, row-major) into one bf16 matrix HID x 2F.
__global__ void convert_wcat(const float* __restrict__ Wl, const float* __restrict__ Wr,
                             unsigned short* __restrict__ out, int F) {
    int K2 = 2 * F;
    long total  = (long)HID_C * K2;
    long stride = (long)gridDim.x * blockDim.x;
    for (long i = (long)blockIdx.x * blockDim.x + threadIdx.x; i < total; i += stride) {
        int r = (int)(i / K2);
        int c = (int)(i - (long)r * K2);
        float v = (c < F) ? Wl[(long)r * F + c] : Wr[(long)r * F + (c - F)];
        out[i] = f2bf(v);
    }
}

// ---------------------------------------------------------------------------
// Edge scatter: msgs[dst] += h[src], 4 floats per thread
// ---------------------------------------------------------------------------
template <int F>
__global__ void scatter_add(const float* __restrict__ h, const int* __restrict__ src,
                            const int* __restrict__ dst, float* __restrict__ msgs) {
    const long total = (long)N_EDGES_C * (F / 4);
    long idx = (long)blockIdx.x * blockDim.x + threadIdx.x;
    if (idx >= total) return;
    int e = (int)(idx / (F / 4));
    int c = (int)(idx % (F / 4)) * 4;
    int s = src[e], d = dst[e];
    const float* hp = h    + (long)s * F + c;
    float*       mp = msgs + (long)d * F + c;
    atomicAdd(mp + 0, hp[0]);
    atomicAdd(mp + 1, hp[1]);
    atomicAdd(mp + 2, hp[2]);
    atomicAdd(mp + 3, hp[3]);
}

// ---------------------------------------------------------------------------
// Fused SAGE GEMM:  out = relu( [msgs/deg | h] @ Wcat^T + bias )
//   A: 32 nodes x (2F) bf16, staged in LDS (padded rows, bank-conflict free)
//   B: Wcat (HID x 2F) bf16, loaded directly (2x b128 per fragment)
//   Each of 8 waves computes 2 row-tiles x 2 col-tiles of 16x16 outputs
//   (cols 16w and 128+16w), sharing B fragments across row-tiles.
// ---------------------------------------------------------------------------
template <int F>
__global__ __launch_bounds__(256) void sage_gemm(
    const float* __restrict__ msgs, const float* __restrict__ deg,
    const float* __restrict__ hprev, const unsigned short* __restrict__ Wcat,
    const float* __restrict__ bias, float* __restrict__ out) {

    constexpr int K2   = 2 * F;
    constexpr int RS   = K2 + 8;               // LDS row stride (bf16 units)
    constexpr int ROWS = 32;                   // nodes per block (2 row-tiles)
    __shared__ unsigned short sA[ROWS * RS];
    __shared__ float sInv[ROWS];

    const int tid  = threadIdx.x;
    const int row0 = blockIdx.x * ROWS;

    if (tid < ROWS) {
        float d = deg[row0 + tid];
        sInv[tid] = 1.0f / fmaxf(d, 1.0f);
    }
    __syncthreads();

    // Stage A = [msgs/deg | hprev] as bf16 pairs into LDS
    for (int p = tid; p < ROWS * F; p += 256) {   // ROWS*F u32-pairs = ROWS x 2F bf16
        int r   = p / F;
        int c2  = p - r * F;
        int col = 2 * c2;
        long i  = row0 + r;
        float a0, a1;
        if (col < F) {
            float inv = sInv[r];
            a0 = msgs[i * F + col] * inv;
            a1 = msgs[i * F + col + 1] * inv;
        } else {
            a0 = hprev[i * F + (col - F)];
            a1 = hprev[i * F + (col - F) + 1];
        }
        unsigned int pk = (unsigned int)f2bf(a0) | ((unsigned int)f2bf(a1) << 16);
        *(unsigned int*)&sA[r * RS + col] = pk;
    }
    __syncthreads();

    const int lane = tid & 31;
    const int wave = tid >> 5;
    const int m    = lane & 15;         // A row / C col within tile
    const int hh   = lane >> 4;         // lane half
    const int n0a  = wave * 16;
    const int n0b  = 128 + wave * 16;

    v8f acc00 = {};   // row-tile 0, col-tile a
    v8f acc01 = {};   // row-tile 0, col-tile b
    v8f acc10 = {};   // row-tile 1, col-tile a
    v8f acc11 = {};   // row-tile 1, col-tile b

    const unsigned short* wa = Wcat + (long)(n0a + m) * K2;
    const unsigned short* wb = Wcat + (long)(n0b + m) * K2;
    const int abase0 = (m)      * RS + 16 * hh;
    const int abase1 = (16 + m) * RS + 16 * hh;

    for (int k0 = 0; k0 < K2; k0 += 32) {
        // B fragments: 16 consecutive bf16 of the weight row, K = k0 + 16*hh + 0..15
        const unsigned short* pb0 = wa + k0 + 16 * hh;
        v4u lo0 = *(const v4u*)pb0;
        v4u hi0 = *(const v4u*)(pb0 + 8);
        v8u bu0; bu0[0]=lo0[0]; bu0[1]=lo0[1]; bu0[2]=lo0[2]; bu0[3]=lo0[3];
                 bu0[4]=hi0[0]; bu0[5]=hi0[1]; bu0[6]=hi0[2]; bu0[7]=hi0[3];
        v16bf bfrag0 = __builtin_bit_cast(v16bf, bu0);

        const unsigned short* pb1 = wb + k0 + 16 * hh;
        v4u lo1 = *(const v4u*)pb1;
        v4u hi1 = *(const v4u*)(pb1 + 8);
        v8u bu1; bu1[0]=lo1[0]; bu1[1]=lo1[1]; bu1[2]=lo1[2]; bu1[3]=lo1[3];
                 bu1[4]=hi1[0]; bu1[5]=hi1[1]; bu1[6]=hi1[2]; bu1[7]=hi1[3];
        v16bf bfrag1 = __builtin_bit_cast(v16bf, bu1);

        // A fragment, row-tile 0: contiguous k order (k = 16*hh + 0..15), same as B.
        // VGPR v: K = 2v + 16*hh  -> one contiguous 32B run in LDS (matches B)
        v8u au0;
#pragma unroll
        for (int v = 0; v < 8; ++v) {
            int koff = 2 * v;
            au0[v] = *(const unsigned int*)&sA[abase0 + k0 + koff];
        }
        v16bf afrag0 = __builtin_bit_cast(v16bf, au0);

        acc00 = __builtin_amdgcn_wmma_f32_16x16x32_bf16(false, afrag0, false, bfrag0,
                                                        (short)0, acc00, false, false);
        acc01 = __builtin_amdgcn_wmma_f32_16x16x32_bf16(false, afrag0, false, bfrag1,
                                                        (short)0, acc01, false, false);

        // A fragment, row-tile 1
        v8u au1;
#pragma unroll
        for (int v = 0; v < 8; ++v) {
            int koff = 2 * v;
            au1[v] = *(const unsigned int*)&sA[abase1 + k0 + koff];
        }
        v16bf afrag1 = __builtin_bit_cast(v16bf, au1);

        acc10 = __builtin_amdgcn_wmma_f32_16x16x32_bf16(false, afrag1, false, bfrag0,
                                                        (short)0, acc10, false, false);
        acc11 = __builtin_amdgcn_wmma_f32_16x16x32_bf16(false, afrag1, false, bfrag1,
                                                        (short)0, acc11, false, false);
    }

    // C/D layout: lane L (col = L&15), VGPR j -> row j + 8*(L>>4)
    const float ba = bias[n0a + m];
    const float bb = bias[n0b + m];
#pragma unroll
    for (int j = 0; j < 8; ++j) {
        int  mrow = j + 8 * hh;
        long o0   = (long)(row0 + mrow) * HID_C;
        long o1   = (long)(row0 + 16 + mrow) * HID_C;
        out[o0 + n0a + m] = fmaxf(acc00[j] + ba, 0.0f);
        out[o0 + n0b + m] = fmaxf(acc01[j] + bb, 0.0f);
        out[o1 + n0a + m] = fmaxf(acc10[j] + ba, 0.0f);
        out[o1 + n0b + m] = fmaxf(acc11[j] + bb, 0.0f);
    }
}

// ---------------------------------------------------------------------------
// Head: pairwise pool (|diff|, mean, max) -> merger -> FF -> log_softmax
// One block per graph, 256 threads.
// ---------------------------------------------------------------------------
__global__ __launch_bounds__(256) void head_kernel(
    const float* __restrict__ h3, const int* __restrict__ base,
    const int* __restrict__ set_idx,
    const float* __restrict__ mW, const float* __restrict__ mB,
    const float* __restrict__ W1, const float* __restrict__ b1,
    const float* __restrict__ W2, const float* __restrict__ b2,
    float* __restrict__ outp) {

    __shared__ float sconc[3 * HID_C];
    __shared__ float spool[HID_C];
    __shared__ float sff[HID_C];
    __shared__ float slog[OUT_F_C];
    __shared__ float sred[2];

    int g = blockIdx.x;
    int t = threadIdx.x;
    int i0 = base[g] + set_idx[g * 2 + 0];
    int i1 = base[g] + set_idx[g * 2 + 1];
    float x0 = h3[(long)i0 * HID_C + t];
    float x1 = h3[(long)i1 * HID_C + t];
    sconc[t]             = fabsf(x0 - x1);
    sconc[HID_C + t]     = 0.5f * (x0 + x1);
    sconc[2 * HID_C + t] = fmaxf(x0, x1);
    __syncthreads();

    float acc = mB[t];
    const float* wr = mW + (long)t * (3 * HID_C);
    for (int j = 0; j < 3 * HID_C; ++j) acc += wr[j] * sconc[j];
    spool[t] = acc;
    __syncthreads();

    float acc2 = b1[t];
    const float* w1r = W1 + (long)t * HID_C;
    for (int j = 0; j < HID_C; ++j) acc2 += w1r[j] * spool[j];
    sff[t] = fmaxf(acc2, 0.0f);
    __syncthreads();

    if (t < OUT_F_C) {
        float a = b2[t];
        const float* w2r = W2 + (long)t * HID_C;
        for (int j = 0; j < HID_C; ++j) a += w2r[j] * sff[j];
        slog[t] = a;
    }
    __syncthreads();

    if (t == 0) {
        float mx = slog[0];
        for (int j = 1; j < OUT_F_C; ++j) mx = fmaxf(mx, slog[j]);
        float s = 0.0f;
        for (int j = 0; j < OUT_F_C; ++j) s += expf(slog[j] - mx);
        sred[0] = mx;
        sred[1] = logf(s);
    }
    __syncthreads();

    if (t < OUT_F_C) outp[(long)g * OUT_F_C + t] = slog[t] - sred[0] - sred[1];
}

// ---------------------------------------------------------------------------
// Launch
// ---------------------------------------------------------------------------
static inline size_t align256(size_t x) { return (x + 255) & ~(size_t)255; }

extern "C" void kernel_launch(void* const* d_in, const int* in_sizes, int n_in,
                              void* d_out, int out_size, void* d_ws, size_t ws_size,
                              hipStream_t stream) {
    (void)in_sizes; (void)n_in; (void)out_size; (void)ws_size;

    const float* x    = (const float*)d_in[0];
    const int*   ei   = (const int*)d_in[1];
    const int*   src  = ei;
    const int*   dst  = ei + N_EDGES_C;
    const int*   bvec = (const int*)d_in[2];
    const int*   sidx = (const int*)d_in[3];
    const float* Wl0 = (const float*)d_in[4];
    const float* bl0 = (const float*)d_in[5];
    const float* Wr0 = (const float*)d_in[6];
    const float* Wl1 = (const float*)d_in[7];
    const float* bl1 = (const float*)d_in[8];
    const float* Wr1 = (const float*)d_in[9];
    const float* Wl2 = (const float*)d_in[10];
    const float* bl2 = (const float*)d_in[11];
    const float* Wr2 = (const float*)d_in[12];
    const float* mW  = (const float*)d_in[13];
    const float* mB  = (const float*)d_in[14];
    const float* W1  = (const float*)d_in[15];
    const float* b1  = (const float*)d_in[16];
    const float* W2  = (const float*)d_in[17];
    const float* b2  = (const float*)d_in[18];

    // Workspace carve-out
    char* ws = (char*)d_ws;
    size_t off = 0;
    float* msgs = (float*)(ws + off); off = align256(off + (size_t)N_NODES_C * HID_C * 4);
    float* deg  = (float*)(ws + off); off = align256(off + (size_t)N_NODES_C * 4);
    float* hA   = (float*)(ws + off); off = align256(off + (size_t)N_NODES_C * HID_C * 4);
    float* hB   = (float*)(ws + off); off = align256(off + (size_t)N_NODES_C * HID_C * 4);
    unsigned short* Wc0 = (unsigned short*)(ws + off); off = align256(off + (size_t)HID_C * (2 * IN_F_C) * 2);
    unsigned short* Wc1 = (unsigned short*)(ws + off); off = align256(off + (size_t)HID_C * (2 * HID_C) * 2);
    unsigned short* Wc2 = (unsigned short*)(ws + off); off = align256(off + (size_t)HID_C * (2 * HID_C) * 2);
    int* cnt  = (int*)(ws + off); off = align256(off + (size_t)N_GRAPHS_C * 4);
    int* base = (int*)(ws + off); off = align256(off + (size_t)N_GRAPHS_C * 4);

    // Degree + per-graph bases (once per launch; deterministic)
    zero_f32<<<2048, 256, 0, stream>>>(deg, N_NODES_C);
    zero_f32<<<64, 256, 0, stream>>>((float*)cnt, N_GRAPHS_C);
    deg_kernel<<<(N_EDGES_C + 255) / 256, 256, 0, stream>>>(dst, deg);
    count_kernel<<<(N_NODES_C + 255) / 256, 256, 0, stream>>>(bvec, cnt);
    scan_bases<<<1, 32, 0, stream>>>(cnt, base);

    // Weight conversion to bf16 [Wl | Wr]
    convert_wcat<<<256, 256, 0, stream>>>(Wl0, Wr0, Wc0, IN_F_C);
    convert_wcat<<<256, 256, 0, stream>>>(Wl1, Wr1, Wc1, HID_C);
    convert_wcat<<<256, 256, 0, stream>>>(Wl2, Wr2, Wc2, HID_C);

    const int tiles = N_NODES_C / 32;   // 3125, exact

    // Layer 0: F = 128, input x -> hA
    zero_f32<<<4096, 256, 0, stream>>>(msgs, (long)N_NODES_C * IN_F_C);
    {
        long tot = (long)N_EDGES_C * (IN_F_C / 4);
        scatter_add<IN_F_C><<<(unsigned)((tot + 255) / 256), 256, 0, stream>>>(x, src, dst, msgs);
    }
    sage_gemm<IN_F_C><<<tiles, 256, 0, stream>>>(msgs, deg, x, Wc0, bl0, hA);

    // Layer 1: F = 256, hA -> hB
    zero_f32<<<4096, 256, 0, stream>>>(msgs, (long)N_NODES_C * HID_C);
    {
        long tot = (long)N_EDGES_C * (HID_C / 4);
        scatter_add<HID_C><<<(unsigned)((tot + 255) / 256), 256, 0, stream>>>(hA, src, dst, msgs);
    }
    sage_gemm<HID_C><<<tiles, 256, 0, stream>>>(msgs, deg, hA, Wc1, bl1, hB);

    // Layer 2: F = 256, hB -> hA
    zero_f32<<<4096, 256, 0, stream>>>(msgs, (long)N_NODES_C * HID_C);
    {
        long tot = (long)N_EDGES_C * (HID_C / 4);
        scatter_add<HID_C><<<(unsigned)((tot + 255) / 256), 256, 0, stream>>>(hB, src, dst, msgs);
    }
    sage_gemm<HID_C><<<tiles, 256, 0, stream>>>(msgs, deg, hB, Wc2, bl2, hA);

    // Head
    head_kernel<<<N_GRAPHS_C, 256, 0, stream>>>(hA, base, sidx, mW, mB, W1, b1, W2, b2,
                                                (float*)d_out);
}
